// HierBertLayer_3702261809251
// MI455X (gfx1250) — compile-verified, matrix operand order checked by reading, not run on hardware
//
#include <hip/hip_runtime.h>
#include <hip/hip_bf16.h>
#include <math.h>

// ---------------------------------------------------------------------------
// HierBertLayer for MI455X (gfx1250, wave32).
// All GEMM-shaped work runs on v_wmma_f32_16x16x32_f16 (f16 in, f32 accum).
// Data movement uses GLOBAL_LOAD_ASYNC_TO_LDS (ASYNCcnt-tracked, direct
// memory->LDS) plus global_prefetch for the next K panel.
// ---------------------------------------------------------------------------

typedef __attribute__((ext_vector_type(16))) _Float16 v16h;
typedef __attribute__((ext_vector_type(8)))  float    v8f;
typedef int v4i __attribute__((vector_size(16)));     // matches builtin param

#define Bb     8
#define Ss     512
#define Hh     768
#define Ff     3072
#define NHEAD  12
#define DHEAD  64
#define Mrows  (Bb * Ss)   // 4096

#if defined(__has_builtin)
#if __has_builtin(__builtin_amdgcn_global_load_async_to_lds_b128) && \
    __has_builtin(__builtin_amdgcn_s_wait_asynccnt)
#define USE_ASYNC_LDS 1
#endif
#endif

union FragH  { v16h v; unsigned int u[8]; };
union Vec16h { float4 f4[2]; _Float16 h[16]; };
union Vec32h { float4 f4[4]; _Float16 h[32]; };

// 16-byte straight copy global -> LDS. Async (no VGPR roundtrip) when available.
__device__ __forceinline__ void copy16B_to_lds(const _Float16* gsrc, _Float16* ldst) {
#if defined(USE_ASYNC_LDS)
    __builtin_amdgcn_global_load_async_to_lds_b128(
        (__attribute__((address_space(1))) v4i*)(void*)gsrc,
        (__attribute__((address_space(3))) v4i*)(void*)ldst, 0, 0);
#else
    *(float4*)ldst = *(const float4*)gsrc;
#endif
}
__device__ __forceinline__ void copy_wait() {
#if defined(USE_ASYNC_LDS)
    __builtin_amdgcn_s_wait_asynccnt(0);
#endif
}

// Load a 16x32 f16 WMMA operand fragment (A layout; B uses the mirrored
// layout, so the same loader works when `base` indexes the N-major tile).
// ISA 16-bit A 16x32: lane L holds row L&15; VGPR i holds K pair
//   kb = (i<4 ? 2i : 16+2(i-4)) + 8*(L>>4), packed low/high.
__device__ __forceinline__ v16h load_frag16(const _Float16* base, int stride, int lane) {
    const int r  = lane & 15;
    const int hh = lane >> 4;
    const _Float16* row = base + r * stride;
    FragH f;
#pragma unroll
    for (int i = 0; i < 8; ++i) {
        int kb = ((i < 4) ? (2 * i) : (16 + 2 * (i - 4))) + 8 * hh;
        f.u[i] = *(const unsigned int*)(row + kb);   // K-pair contiguous -> one b32
    }
    return f.v;
}

#define WMMA_F16(A, B, C) \
    __builtin_amdgcn_wmma_f32_16x16x32_f16(false, (A), false, (B), (short)0, (C), false, false)

// ---------------------------------------------------------------------------
// Tiled WMMA GEMM: Y[M,N] = act(X[M,K] @ W[K,N] + bias[N])
// Block = 128 threads (4 waves, 2Mx2N), block tile 64x128, wave tile 32x64,
// K-step 32 -> 8 WMMAs per wave per K-step.
// ---------------------------------------------------------------------------
template <bool GELU, bool OUT16>
__global__ void __launch_bounds__(128) gemm_kernel(
        const _Float16* __restrict__ X, const _Float16* __restrict__ W,
        const float* __restrict__ bias, void* __restrict__ Y,
        int M, int N, int K) {
    constexpr int LDA = 40;                       // padded stride (halfs)
    __shared__ _Float16 At[64 * LDA];             // A tile [row][k]
    __shared__ _Float16 Bt[128 * LDA];            // B tile transposed [col][k]
    const int tid  = threadIdx.x;
    const int lane = tid & 31;
    const int wave = tid >> 5;
    const int m0 = blockIdx.y * 64;
    const int n0 = blockIdx.x * 128;
    const int wm = (wave >> 1) * 32;
    const int wn = (wave & 1) * 64;

    v8f acc[2][4] = {};

    for (int k0 = 0; k0 < K; k0 += 32) {
        {   // stage A: 64 rows x 32 k  (16 halfs per thread, 2x 16B)
            int r  = tid >> 1;
            int cb = (tid & 1) * 16;
            const _Float16* src = X + (size_t)(m0 + r) * K + k0 + cb;
            _Float16* dst = &At[r * LDA + cb];
            copy16B_to_lds(src,     dst);
            copy16B_to_lds(src + 8, dst + 8);
            if (k0 + 32 < K) __builtin_prefetch(src + 32, 0, 0);
        }
        {   // stage B transposed: W rows k0..k0+31, cols n0..n0+127 -> Bt[col][k]
            int kk   = tid >> 2;
            int cseg = (tid & 3) * 32;
            Vec32h tmp;
            const float4* src = (const float4*)(W + (size_t)(k0 + kk) * N + n0 + cseg);
#pragma unroll
            for (int j = 0; j < 4; ++j) tmp.f4[j] = src[j];
#pragma unroll
            for (int j = 0; j < 32; ++j) Bt[(cseg + j) * LDA + kk] = tmp.h[j];
            if (k0 + 32 < K) __builtin_prefetch((const _Float16*)src + (size_t)32 * N, 0, 0);
        }
        copy_wait();
        __syncthreads();

        v16h a0 = load_frag16(&At[(wm +  0) * LDA], LDA, lane);
        v16h a1 = load_frag16(&At[(wm + 16) * LDA], LDA, lane);
#pragma unroll
        for (int tj = 0; tj < 4; ++tj) {
            v16h b = load_frag16(&Bt[(wn + tj * 16) * LDA], LDA, lane);
            acc[0][tj] = WMMA_F16(a0, b, acc[0][tj]);
            acc[1][tj] = WMMA_F16(a1, b, acc[1][tj]);
        }
        __syncthreads();
    }

    const int c  = lane & 15;
    const int hh = lane >> 4;
#pragma unroll
    for (int ti = 0; ti < 2; ++ti)
#pragma unroll
    for (int tj = 0; tj < 4; ++tj) {
        int col = n0 + wn + tj * 16 + c;
        float bc = bias[col];
#pragma unroll
        for (int v = 0; v < 8; ++v) {
            int row = m0 + wm + ti * 16 + v + 8 * hh;
            float val = acc[ti][tj][v] + bc;
            if (GELU) val = 0.5f * val * (1.0f + erff(val * 0.70710678118654752f));
            if (OUT16) ((_Float16*)Y)[(size_t)row * N + col] = (_Float16)val;
            else       ((float*)Y)[(size_t)row * N + col]    = val;
        }
    }
}

// ---------------------------------------------------------------------------
// Attention core. Block = 64 threads (2 waves); each wave owns a 16-query
// tile of one (batch, head). Additive mask row precomputed in LDS (branch-
// free inner loop). Scores per-wave in LDS (16x512 f32), softmax, f16 P,
// ctx = P @ V via WMMA.   gid==0: ext mask; gid>=1: hier group mask.
// ---------------------------------------------------------------------------
#define SMEM_ATTN (2*16*Ss*4 + 2*16*Ss*2 + 16*DHEAD*2 + DHEAD*32*2 + Ss*4)  // 106496 B

__global__ void __launch_bounds__(64) attn_kernel(
        const _Float16* __restrict__ Q, const _Float16* __restrict__ Km,
        const _Float16* __restrict__ V, const float* __restrict__ extmask,
        const int* __restrict__ hmask, int gid, _Float16* __restrict__ Ctx) {
    extern __shared__ char smem[];
    float*    scores = (float*)smem;                                   // [2][16][512]
    _Float16* pbuf   = (_Float16*)(smem + 2*16*Ss*4);                  // [2][16][512]
    _Float16* kst    = (_Float16*)(smem + 2*16*Ss*4 + 2*16*Ss*2);      // [16][64]
    _Float16* vst    = kst + 16*DHEAD;                                 // [64][32] (dh-major)
    float*    mrow   = (float*)(vst + DHEAD*32);                       // [512]

    const int tid  = threadIdx.x;
    const int lane = tid & 31;
    const int wave = tid >> 5;
    const int blk  = blockIdx.x;
    const int qt   = blk & 15;                 // 32-query chunk within S
    const int head = (blk >> 4) % NHEAD;
    const int b    = blk / (16 * NHEAD);
    const int q0   = qt * 32 + wave * 16;

    // precompute additive mask row (hoists all branching out of the hot loop)
#pragma unroll
    for (int j = tid; j < Ss; j += 64)
        mrow[j] = (gid == 0) ? extmask[b * Ss + j]
                             : ((hmask[b * Ss + j] == gid) ? 0.0f : -10000.0f);

    // Q fragments (16 queries x 64 dh, two 16x32 chunks)
    const _Float16* qbase = Q + (size_t)(b * Ss + q0) * Hh + head * DHEAD;
    v16h qa0 = load_frag16(qbase +  0, Hh, lane);
    v16h qa1 = load_frag16(qbase + 32, Hh, lane);

    float* swave = scores + wave * 16 * Ss;
    __syncthreads();

    // ---- phase 1: masked scores -------------------------------------------
    for (int kt = 0; kt < Ss / 16; ++kt) {
        {   // stage 16 keys x 64 dh (row-major; straight copy -> async path)
            int key  = tid >> 2;
            int cseg = (tid & 3) * 16;
            const _Float16* src = Km + (size_t)(b*Ss + kt*16 + key) * Hh + head*DHEAD + cseg;
            _Float16* dst = kst + key * DHEAD + cseg;
            copy16B_to_lds(src,     dst);
            copy16B_to_lds(src + 8, dst + 8);
        }
        copy_wait();
        __syncthreads();
        v8f sacc = {};
        v16h kb0 = load_frag16(kst +  0, DHEAD, lane);   // B = K^T chunk 0
        v16h kb1 = load_frag16(kst + 32, DHEAD, lane);   // B = K^T chunk 1
        sacc = WMMA_F16(qa0, kb0, sacc);
        sacc = WMMA_F16(qa1, kb1, sacc);

        const int c    = lane & 15;
        const int hh   = lane >> 4;
        const int kpos = kt * 16 + c;
        const float mval = mrow[kpos];
#pragma unroll
        for (int v = 0; v < 8; ++v)
            swave[(v + 8 * hh) * Ss + kpos] = sacc[v] * 0.125f + mval;  // 1/sqrt(64)
        __syncthreads();
    }

    // ---- phase 2: softmax (lane pair l / l+16 splits each row) ------------
    {
        const int r  = lane & 15;
        const int hh = lane >> 4;
        float* rowp = swave + r * Ss + hh * 256;
        float mx = -3.0e38f;
        for (int j = 0; j < 256; ++j) mx = fmaxf(mx, rowp[j]);
        mx = fmaxf(mx, __shfl_xor(mx, 16, 32));
        float sum = 0.0f;
        for (int j = 0; j < 256; ++j) { float e = __expf(rowp[j] - mx); rowp[j] = e; sum += e; }
        sum += __shfl_xor(sum, 16, 32);
        float inv = 1.0f / sum;
        _Float16* prow = pbuf + wave * 16 * Ss + r * Ss + hh * 256;
        for (int j = 0; j < 256; ++j) prow[j] = (_Float16)(rowp[j] * inv);
    }
    __syncthreads();

    // ---- phase 3: ctx = P(16x512) @ V(512x64) -----------------------------
    v8f cacc[4] = {};
    for (int kk = 0; kk < Ss / 32; ++kk) {
        {   // stage V chunk transposed: vst[dh][key] so key-pairs are contiguous
            int key  = tid >> 1;
            int cseg = (tid & 1) * 32;
            Vec32h tmp;
            const float4* src = (const float4*)(V + (size_t)(b*Ss + kk*32 + key) * Hh + head*DHEAD + cseg);
#pragma unroll
            for (int j = 0; j < 4; ++j) tmp.f4[j] = src[j];
#pragma unroll
            for (int j = 0; j < 32; ++j) vst[(cseg + j) * 32 + key] = tmp.h[j];
        }
        __syncthreads();
        v16h pa = load_frag16(pbuf + wave * 16 * Ss + kk * 32, Ss, lane);
#pragma unroll
        for (int nt = 0; nt < 4; ++nt) {
            v16h vb = load_frag16(vst + nt * 16 * 32, 32, lane);
            cacc[nt] = WMMA_F16(pa, vb, cacc[nt]);
        }
        __syncthreads();
    }

    const int c  = lane & 15;
    const int hh = lane >> 4;
#pragma unroll
    for (int nt = 0; nt < 4; ++nt)
#pragma unroll
    for (int v = 0; v < 8; ++v)
        Ctx[(size_t)(b*Ss + q0 + v + 8*hh) * Hh + head*DHEAD + nt*16 + c] = (_Float16)cacc[nt][v];
}

// ---------------------------------------------------------------------------
// LayerNorm(x + resid) over H=768, block = 256 threads (3 cols each).
// Optional f32 / f16 outputs; optional gacc += y * (hmask[row]==gid).
// ---------------------------------------------------------------------------
__global__ void __launch_bounds__(256) ln_kernel(
        const float* __restrict__ x, const float* __restrict__ resid,
        const float* __restrict__ lnw, float* __restrict__ out32,
        _Float16* __restrict__ out16, float* __restrict__ gacc,
        const int* __restrict__ hmask, int gid) {
    __shared__ float red[256];
    const int row = blockIdx.x;
    const int t   = threadIdx.x;
    const float* xr = x + (size_t)row * Hh;
    const float* rr = resid + (size_t)row * Hh;
    float v[3];
    float s = 0.0f;
#pragma unroll
    for (int i = 0; i < 3; ++i) { int c = t * 3 + i; v[i] = xr[c] + rr[c]; s += v[i]; }
    red[t] = s; __syncthreads();
    for (int st = 128; st > 0; st >>= 1) { if (t < st) red[t] += red[t + st]; __syncthreads(); }
    const float mean = red[0] * (1.0f / Hh);
    __syncthreads();
    s = 0.0f;
#pragma unroll
    for (int i = 0; i < 3; ++i) { float d = v[i] - mean; s += d * d; }
    red[t] = s; __syncthreads();
    for (int st = 128; st > 0; st >>= 1) { if (t < st) red[t] += red[t + st]; __syncthreads(); }
    const float inv = rsqrtf(red[0] * (1.0f / Hh) + 1e-12f);
    float mk = 0.0f;
    if (gacc) mk = (hmask[row] == gid) ? 1.0f : 0.0f;
#pragma unroll
    for (int i = 0; i < 3; ++i) {
        int c = t * 3 + i;
        float y = (v[i] - mean) * inv * lnw[c] + lnw[Hh + c];
        if (out32) out32[(size_t)row * Hh + c] = y;
        if (out16) out16[(size_t)row * Hh + c] = (_Float16)y;
        if (gacc)  gacc[(size_t)row * Hh + c] += y * mk;
    }
}

// ---------------------------------------------------------------------------
// Elementwise helpers
// ---------------------------------------------------------------------------
__global__ void cast_f16_kernel(const float* __restrict__ s, _Float16* __restrict__ d, int n) {
    int i = blockIdx.x * 256 + threadIdx.x;
    if (i < n) d[i] = (_Float16)s[i];
}
__global__ void zero_kernel(float* __restrict__ p, int n) {
    int i = blockIdx.x * 256 + threadIdx.x;
    if (i < n) p[i] = 0.0f;
}
__global__ void combine_kernel(const float* __restrict__ a, const float* __restrict__ b,
                               float* __restrict__ o32, _Float16* __restrict__ o16, int n) {
    int i = blockIdx.x * 256 + threadIdx.x;
    if (i < n) { float v = a[i] + b[i]; o32[i] = v; o16[i] = (_Float16)v; }
}

// ---------------------------------------------------------------------------
extern "C" void kernel_launch(void* const* d_in, const int* in_sizes, int n_in,
                              void* d_out, int out_size, void* d_ws, size_t ws_size,
                              hipStream_t stream) {
    (void)in_sizes; (void)n_in; (void)out_size; (void)ws_size;

    const float* h32    = (const float*)d_in[0];
    const float* amask  = (const float*)d_in[1];   // [B,1,1,S]
    const int*   hmask  = (const int*)d_in[2];     // [B,S]
    const float* mWattn = (const float*)d_in[3];   // [4,H,H]
    const float* mbattn = (const float*)d_in[4];   // [4,H]
    const float* mlnA   = (const float*)d_in[5];   // [2,H]
    const float* mWi    = (const float*)d_in[6];
    const float* mbi    = (const float*)d_in[7];
    const float* mWo    = (const float*)d_in[8];
    const float* mbo    = (const float*)d_in[9];
    const float* mlnO   = (const float*)d_in[10];
    const float* hWattn = (const float*)d_in[11];
    const float* hbattn = (const float*)d_in[12];
    const float* hlnA   = (const float*)d_in[13];
    const float* hWi    = (const float*)d_in[14];
    const float* hbi    = (const float*)d_in[15];
    const float* hWo    = (const float*)d_in[16];
    const float* hbo    = (const float*)d_in[17];
    const float* hlnO   = (const float*)d_in[18];

    char* wsb = (char*)d_ws;
    size_t off = 0;
    auto alloc = [&](size_t bytes) -> char* {
        char* p = wsb + off; off += (bytes + 255) & ~(size_t)255; return p;
    };
    _Float16* h16    = (_Float16*)alloc((size_t)Mrows * Hh * 2);
    _Float16* mA16   = (_Float16*)alloc((size_t)4 * Hh * Hh * 2);
    _Float16* mWi16  = (_Float16*)alloc((size_t)Hh * Ff * 2);
    _Float16* mWo16  = (_Float16*)alloc((size_t)Ff * Hh * 2);
    _Float16* hA16   = (_Float16*)alloc((size_t)4 * Hh * Hh * 2);
    _Float16* hWi16  = (_Float16*)alloc((size_t)Hh * Ff * 2);
    _Float16* hWo16  = (_Float16*)alloc((size_t)Ff * Hh * 2);
    _Float16* q16    = (_Float16*)alloc((size_t)Mrows * Hh * 2);
    _Float16* k16    = (_Float16*)alloc((size_t)Mrows * Hh * 2);
    _Float16* v16    = (_Float16*)alloc((size_t)Mrows * Hh * 2);
    _Float16* hq16   = (_Float16*)alloc((size_t)Mrows * Hh * 2);
    _Float16* hk16   = (_Float16*)alloc((size_t)Mrows * Hh * 2);
    _Float16* hv16   = (_Float16*)alloc((size_t)Mrows * Hh * 2);
    _Float16* ctx16  = (_Float16*)alloc((size_t)Mrows * Hh * 2);
    float*    acc32  = (float*)alloc((size_t)Mrows * Hh * 4);
    float*    mAtt32 = (float*)alloc((size_t)Mrows * Hh * 4);
    _Float16* a16    = (_Float16*)alloc((size_t)Mrows * Hh * 2);
    float*    a32    = (float*)alloc((size_t)Mrows * Hh * 4);
    _Float16* int16b = (_Float16*)alloc((size_t)Mrows * Ff * 2);
    float*    hAcc   = (float*)alloc((size_t)Mrows * Hh * 4);
    float*    comb32 = (float*)alloc((size_t)Mrows * Hh * 4);
    _Float16* comb16 = (_Float16*)alloc((size_t)Mrows * Hh * 2);

    auto cast = [&](const float* s, _Float16* d, int n) {
        cast_f16_kernel<<<(n + 255) / 256, 256, 0, stream>>>(s, d, n);
    };
    auto gemm16 = [&](const _Float16* X, const _Float16* W, const float* b, _Float16* Y,
                      int M, int N, int K) {
        gemm_kernel<false, true><<<dim3(N / 128, M / 64), 128, 0, stream>>>(X, W, b, (void*)Y, M, N, K);
    };
    auto gemm32 = [&](const _Float16* X, const _Float16* W, const float* b, float* Y,
                      int M, int N, int K) {
        gemm_kernel<false, false><<<dim3(N / 128, M / 64), 128, 0, stream>>>(X, W, b, (void*)Y, M, N, K);
    };
    auto gemmGelu = [&](const _Float16* X, const _Float16* W, const float* b, _Float16* Y,
                        int M, int N, int K) {
        gemm_kernel<true, true><<<dim3(N / 128, M / 64), 128, 0, stream>>>(X, W, b, (void*)Y, M, N, K);
    };

    // -- casts ---------------------------------------------------------------
    cast(h32,    h16,   Mrows * Hh);
    cast(mWattn, mA16,  4 * Hh * Hh);
    cast(mWi,    mWi16, Hh * Ff);
    cast(mWo,    mWo16, Ff * Hh);
    cast(hWattn, hA16,  4 * Hh * Hh);
    cast(hWi,    hWi16, Hh * Ff);
    cast(hWo,    hWo16, Ff * Hh);
    zero_kernel<<<(Mrows * Hh + 255) / 256, 256, 0, stream>>>(hAcc, Mrows * Hh);

    const int attnGrid = Bb * NHEAD * (Ss / 32);   // 1536 blocks of 64 threads

    // -- main attention -------------------------------------------------------
    gemm16(h16, mA16 + 0 * Hh * Hh, mbattn + 0 * Hh, q16, Mrows, Hh, Hh);
    gemm16(h16, mA16 + 1 * Hh * Hh, mbattn + 1 * Hh, k16, Mrows, Hh, Hh);
    gemm16(h16, mA16 + 2 * Hh * Hh, mbattn + 2 * Hh, v16, Mrows, Hh, Hh);
    attn_kernel<<<attnGrid, 64, SMEM_ATTN, stream>>>(q16, k16, v16, amask, hmask, 0, ctx16);
    gemm32(ctx16, mA16 + 3 * Hh * Hh, mbattn + 3 * Hh, acc32, Mrows, Hh, Hh);
    ln_kernel<<<Mrows, 256, 0, stream>>>(acc32, h32, mlnA, mAtt32, nullptr, nullptr, nullptr, 0);

    // -- hier: QKV shared across groups ---------------------------------------
    gemm16(h16, hA16 + 0 * Hh * Hh, hbattn + 0 * Hh, hq16, Mrows, Hh, Hh);
    gemm16(h16, hA16 + 1 * Hh * Hh, hbattn + 1 * Hh, hk16, Mrows, Hh, Hh);
    gemm16(h16, hA16 + 2 * Hh * Hh, hbattn + 2 * Hh, hv16, Mrows, Hh, Hh);
    for (int g = 1; g <= 4; ++g) {
        attn_kernel<<<attnGrid, 64, SMEM_ATTN, stream>>>(hq16, hk16, hv16, amask, hmask, g, ctx16);
        gemm32(ctx16, hA16 + 3 * Hh * Hh, hbattn + 3 * Hh, acc32, Mrows, Hh, Hh);
        ln_kernel<<<Mrows, 256, 0, stream>>>(acc32, h32, hlnA, a32, a16, nullptr, nullptr, 0);
        gemmGelu(a16, hWi16, hbi, int16b, Mrows, Ff, Hh);
        gemm32(int16b, hWo16, hbo, acc32, Mrows, Hh, Ff);
        ln_kernel<<<Mrows, 256, 0, stream>>>(acc32, a32, hlnO, nullptr, nullptr, hAcc, hmask, g);
    }

    // -- combined + final FFN -------------------------------------------------
    combine_kernel<<<(Mrows * Hh + 255) / 256, 256, 0, stream>>>(mAtt32, hAcc, comb32, comb16, Mrows * Hh);
    gemmGelu(comb16, mWi16, mbi, int16b, Mrows, Ff, Hh);
    gemm32(int16b, mWo16, mbo, acc32, Mrows, Hh, Ff);
    ln_kernel<<<Mrows, 256, 0, stream>>>(acc32, comb32, mlnO, (float*)d_out, nullptr, nullptr, nullptr, 0);
}
